// KNN_itc_11338713662052
// MI455X (gfx1250) — matrix operand order repaired and checked by reading, not compiled, not run on hardware
//
#include <hip/hip_runtime.h>
#include <hip/hip_bf16.h>
#include <math.h>

// ---------------------------------------------------------------------------
// Problem constants (from reference): q [75,640,21,21], S [5,640,2205], av=5
// ---------------------------------------------------------------------------
#define BQ     75
#define CC     640
#define HW     441      // 21*21
#define HWP    448      // padded to 28*16
#define NWAY   5
#define SHOTS  5
#define CHUNK  441
#define CHUNKP 448
#define AV     5
#define GRP    (BQ / AV)        // 15
#define RT     4                // row tiles (of 16) per block: 64 query rows
#define ROWS_PER_BLK (RT * 16)  // 64
#define NEG_INF (-3.0e38f)

typedef __attribute__((ext_vector_type(16))) _Float16 v16h;
typedef __attribute__((ext_vector_type(8)))  _Float16 v8h;
typedef __attribute__((ext_vector_type(8)))  float    v8f;

// ---------------------------------------------------------------------------
// Kernel 1: L2-normalize query descriptors over channel dim.
// q layout [B][C][hw] (channel-major) -> qn [B][HWP][C] f16, padded rows = 0.
// ---------------------------------------------------------------------------
__global__ __launch_bounds__(256) void normalize_q_kernel(
    const float* __restrict__ q, _Float16* __restrict__ qn) {
  int idx = blockIdx.x * blockDim.x + threadIdx.x;
  if (idx >= BQ * HWP) return;
  int b = idx / HWP;
  int x = idx % HWP;
  _Float16* dst = qn + ((size_t)b * HWP + x) * CC;
  if (x < HW) {
    const float* src = q + (size_t)b * CC * HW + x;
    float ss = 0.f;
    for (int c = 0; c < CC; ++c) {
      float v = src[(size_t)c * HW];
      ss += v * v;
    }
    float scale = 1.f / (sqrtf(ss) + 1e-8f);
    for (int c = 0; c < CC; ++c) {
      dst[c] = (_Float16)(src[(size_t)c * HW] * scale);
    }
  } else {
    for (int c = 0; c < CC; ++c) dst[c] = (_Float16)0.f;
  }
}

// ---------------------------------------------------------------------------
// Kernel 2: L2-normalize support descriptors over channel dim.
// S layout [nway][C][M] -> sn [nway][shots][CHUNKP][C] f16, padded cols = 0.
// ---------------------------------------------------------------------------
__global__ __launch_bounds__(256) void normalize_s_kernel(
    const float* __restrict__ S, _Float16* __restrict__ sn) {
  int idx = blockIdx.x * blockDim.x + threadIdx.x;
  if (idx >= NWAY * SHOTS * CHUNKP) return;
  int n   = idx / (SHOTS * CHUNKP);
  int rem = idx % (SHOTS * CHUNKP);
  int s   = rem / CHUNKP;
  int col = rem % CHUNKP;
  _Float16* dst = sn + ((size_t)(n * SHOTS + s) * CHUNKP + col) * CC;
  if (col < CHUNK) {
    int m = s * CHUNK + col;
    const float* src = S + (size_t)n * CC * (SHOTS * CHUNK) + m;
    float ss = 0.f;
    for (int c = 0; c < CC; ++c) {
      float v = src[(size_t)c * (SHOTS * CHUNK)];
      ss += v * v;
    }
    float scale = 1.f / (sqrtf(ss) + 1e-8f);
    for (int c = 0; c < CC; ++c) {
      dst[c] = (_Float16)(src[(size_t)c * (SHOTS * CHUNK)] * scale);
    }
  } else {
    for (int c = 0; c < CC; ++c) dst[c] = (_Float16)0.f;
  }
}

// Branchless insert of v into the sorted triple (t0 >= t1 >= t2): 5 VALU ops,
// no exec-mask branches.
__device__ __forceinline__ void top3_insert(float& t0, float& t1, float& t2, float v) {
  float m0 = fmaxf(t0, v);
  float n0 = fminf(t0, v);
  float m1 = fmaxf(t1, n0);
  float n1 = fminf(t1, n0);
  float m2 = fmaxf(t2, n1);
  t0 = m0; t1 = m1; t2 = m2;
}

// ---------------------------------------------------------------------------
// Kernel 3: fused WMMA GEMM + per-chunk top-3 epilogue.
// grid = (7 row groups of 64 rows, nway, B). block = 160 threads = 5 waves.
// Wave s owns support chunk s. Loop order ct -> k -> rt. Both operands are
// software-pipelined one step ahead: B (global, 2x b128) one K-step ahead,
// A (LDS, 2x ds b128) one (r,k)-step ahead, so each WMMA issues with only
// the prefetch loads outstanding instead of draining loadcnt/dscnt to 0.
// The LDS offset is laundered through inline asm inside the ct loop so LICM
// cannot hoist the (ct-invariant) A fragments into 600+ VGPRs and spill.
// ---------------------------------------------------------------------------
__global__ __launch_bounds__(160) void knn_main_kernel(
    const _Float16* __restrict__ qn, const _Float16* __restrict__ sn,
    float* __restrict__ inner_sim, float* __restrict__ chunk_sum) {
  const int rtg = blockIdx.x;   // row group (64 query positions)
  const int n   = blockIdx.y;   // class
  const int b   = blockIdx.z;   // query image
  const int tid = threadIdx.x;
  const int wave    = tid >> 5;
  const int lane    = tid & 31;
  const int laneCol = lane & 15;
  const int laneHi  = lane >> 4;
  const int rowBase = rtg * ROWS_PER_BLK;

  __shared__ alignas(16) _Float16 ldsA[ROWS_PER_BLK * CC];  // 80 KB
  __shared__ float topLds[SHOTS][16][16][3];                // 15 KB (reused per rt phase)
  __shared__ float chunkTop[SHOTS][16][3];
  __shared__ float chunkRowSum[SHOTS][16];
  __shared__ float rowGlob[16];

  // ---- Stage the shared A tile (64 rows x 640 K, f16) into LDS ----
  {
    const uint4* gsrc = (const uint4*)(qn + ((size_t)b * HWP + rowBase) * CC);
    uint4* ldst = (uint4*)ldsA;
    #pragma unroll
    for (int i = 0; i < (ROWS_PER_BLK * CC / 8) / 160; ++i)
      ldst[i * 160 + tid] = gsrc[i * 160 + tid];
  }
  __syncthreads();

  const int s = wave;  // chunk id (exactly 5 waves)

  // Per-lane running top-3 for 8 rows of each of the RT C/D fragments.
  float t0[RT][8], t1[RT][8], t2[RT][8];
  #pragma unroll
  for (int r = 0; r < RT; ++r)
    #pragma unroll
    for (int i = 0; i < 8; ++i) { t0[r][i] = NEG_INF; t1[r][i] = NEG_INF; t2[r][i] = NEG_INF; }

  {
    // A fragment base offset: lane holds row M = laneCol; half selects K phase.
    const unsigned apOffBase = (unsigned)laneCol * CC + (laneHi ? 16u : 0u);
    const _Float16* Bbase = sn + (size_t)(n * SHOTS + s) * CHUNKP * CC;

    for (int ct = 0; ct < 28; ++ct) {
      const int col = ct * 16 + laneCol;          // chunk-local column
      // B fragment: lane holds column N = laneCol; half selects K 0..15 / 16..31.
      const _Float16* bp = Bbase + (size_t)col * CC + (laneHi ? 16 : 0);
      if (ct + 1 < 28) __builtin_prefetch(bp + 16 * CC, 0, 1);  // next strip

      // Launder the LDS offset so A loads are re-issued (from LDS) each ct
      // iteration instead of being hoisted into registers across the loop.
      unsigned apOff = apOffBase;
      asm volatile("" : "+v"(apOff));

      // A fragment loader (2x ds_load_b128 -> one 16-reg operand).
      auto loadA = [&](unsigned off) -> v16h {
        v8h lo = *(const v8h*)(ldsA + off);
        v8h hi = *(const v8h*)(ldsA + off + 8);
        return __builtin_shufflevector(lo, hi,
                 0, 1, 2, 3, 4, 5, 6, 7, 8, 9, 10, 11, 12, 13, 14, 15);
      };

      v8f c[RT];
      #pragma unroll
      for (int r = 0; r < RT; ++r) c[r] = (v8f){};

      // Prime both pipelines.
      v16h bcur = *(const v16h*)(bp);
      v16h apre = loadA(apOff);

      #pragma unroll
      for (int k = 0; k < CC; k += 32) {
        v16h bnxt = bcur;
        if (k + 32 < CC) bnxt = *(const v16h*)(bp + k + 32);
        #pragma unroll
        for (int r = 0; r < RT; ++r) {
          // Prefetch the A fragment for the next (r,k) step.
          int nr = r + 1, nk = k;
          if (nr == RT) { nr = 0; nk = k + 32; }
          v16h anxt = apre;
          if (nk < CC) anxt = loadA(apOff + (unsigned)(nr * 16 * CC + nk));
          c[r] = __builtin_amdgcn_wmma_f32_16x16x32_f16(
                     false, apre, false, bcur, (short)0, c[r], false, false);
          apre = anxt;
        }
        bcur = bnxt;
      }

      const bool colValid = (col < CHUNK);
      #pragma unroll
      for (int r = 0; r < RT; ++r)
        #pragma unroll
        for (int i = 0; i < 8; ++i) {
          float v = colValid ? c[r][i] : NEG_INF;
          top3_insert(t0[r][i], t1[r][i], t2[r][i], v);
        }
    }
  }

  // ---- Epilogue: 4 sequential phases, one per row tile, reusing topLds ----
  #pragma unroll
  for (int rp = 0; rp < RT; ++rp) {
    __syncthreads();
    #pragma unroll
    for (int i = 0; i < 8; ++i) {
      int row = laneHi * 8 + i;
      topLds[s][row][laneCol][0] = t0[rp][i];
      topLds[s][row][laneCol][1] = t1[rp][i];
      topLds[s][row][laneCol][2] = t2[rp][i];
    }
    __syncthreads();

    // Merge 16 lane-local top-3s -> chunk top-3 per (s,row)
    if (tid < SHOTS * 16) {
      int s2 = tid >> 4, row = tid & 15;
      float a0 = NEG_INF, a1 = NEG_INF, a2 = NEG_INF;
      for (int l = 0; l < 16; ++l) {
        top3_insert(a0, a1, a2, topLds[s2][row][l][0]);
        top3_insert(a0, a1, a2, topLds[s2][row][l][1]);
        top3_insert(a0, a1, a2, topLds[s2][row][l][2]);
      }
      chunkTop[s2][row][0] = a0;
      chunkTop[s2][row][1] = a1;
      chunkTop[s2][row][2] = a2;
      chunkRowSum[s2][row] = a0 + a1 + a2;
    }
    __syncthreads();

    // Global top-3 per row = top-3 of the 5 chunk top-3s
    if (tid < 16) {
      float a0 = NEG_INF, a1 = NEG_INF, a2 = NEG_INF;
      for (int s2 = 0; s2 < SHOTS; ++s2) {
        top3_insert(a0, a1, a2, chunkTop[s2][tid][0]);
        top3_insert(a0, a1, a2, chunkTop[s2][tid][1]);
        top3_insert(a0, a1, a2, chunkTop[s2][tid][2]);
      }
      rowGlob[tid] = a0 + a1 + a2;
    }
    __syncthreads();

    int rowsValid = HW - (rowBase + rp * 16);
    rowsValid = rowsValid < 0 ? 0 : (rowsValid > 16 ? 16 : rowsValid);
    if (tid < SHOTS) {
      float acc = 0.f;
      for (int r = 0; r < rowsValid; ++r) acc += chunkRowSum[tid][r];
      atomicAdd(&chunk_sum[((size_t)b * NWAY + n) * SHOTS + tid], acc);
    } else if (tid == SHOTS) {
      float acc = 0.f;
      for (int r = 0; r < rowsValid; ++r) acc += rowGlob[r];
      atomicAdd(&inner_sim[(size_t)b * NWAY + n], acc);
    }
  }
}

// ---------------------------------------------------------------------------
// Kernel 4: geometric mean over the 5 augmented views.
// out = [sim (15*5)] ++ [tks (15*5*5)]
// ---------------------------------------------------------------------------
__global__ __launch_bounds__(512) void knn_final_kernel(
    const float* __restrict__ inner_sim, const float* __restrict__ chunk_sum,
    float* __restrict__ out) {
  int tid = threadIdx.x;
  if (tid < GRP * NWAY) {
    int g = tid / NWAY, n = tid % NWAY;
    float acc = 0.f;
    for (int a = 0; a < AV; ++a)
      acc += logf(inner_sim[(size_t)(g * AV + a) * NWAY + n]);
    out[tid] = expf(acc * (1.f / AV));
  }
  if (tid < GRP * NWAY * SHOTS) {
    int g = tid / (NWAY * SHOTS);
    int r = tid % (NWAY * SHOTS);
    int n = r / SHOTS, s = r % SHOTS;
    float acc = 0.f;
    for (int a = 0; a < AV; ++a)
      acc += logf(fmaxf(chunk_sum[((size_t)(g * AV + a) * NWAY + n) * SHOTS + s], 1e-8f));
    out[GRP * NWAY + tid] = expf(acc * (1.f / AV));
  }
}

// ---------------------------------------------------------------------------
// Workspace layout (bytes):
//   [0, 43008000)            qn  : f16 [75][448][640]
//   [43008000, 57344000)     sn  : f16 [5][5][448][640]
//   [57344000, 57345536)     inner_sim : f32 [75][5]   (padded)
//   [57345536, 57353036)     chunk_sum : f32 [75][5][5]
// ---------------------------------------------------------------------------
extern "C" void kernel_launch(void* const* d_in, const int* in_sizes, int n_in,
                              void* d_out, int out_size, void* d_ws, size_t ws_size,
                              hipStream_t stream) {
  const float* q = (const float*)d_in[0];
  const float* S = (const float*)d_in[1];
  (void)in_sizes; (void)n_in; (void)out_size; (void)ws_size;

  char* ws = (char*)d_ws;
  _Float16* qn        = (_Float16*)(ws);
  _Float16* sn        = (_Float16*)(ws + 43008000);
  float*    inner_sim = (float*)   (ws + 57344000);
  float*    chunk_sum = (float*)   (ws + 57345536);

  // Zero the accumulators (capture-safe).
  hipMemsetAsync(ws + 57344000, 0, 16384, stream);

  {
    int total = BQ * HWP;                       // 33600
    normalize_q_kernel<<<(total + 255) / 256, 256, 0, stream>>>(q, qn);
  }
  {
    int total = NWAY * SHOTS * CHUNKP;          // 11200
    normalize_s_kernel<<<(total + 255) / 256, 256, 0, stream>>>(S, sn);
  }
  {
    dim3 grid(HWP / ROWS_PER_BLK, NWAY, BQ);    // (7, 5, 75)
    knn_main_kernel<<<grid, 160, 0, stream>>>(qn, sn, inner_sim, chunk_sum);
  }
  knn_final_kernel<<<1, 512, 0, stream>>>(inner_sim, chunk_sum, (float*)d_out);
}
